// Model_7249904795839
// MI455X (gfx1250) — matrix operand order repaired and checked by reading, not compiled, not run on hardware
//
#include <hip/hip_runtime.h>
#include <hip/hip_bf16.h>

#define VOCAB 128
#define EMB 256
#define UNITS 1024
#define BATCH 64
#define SEQLEN 512

typedef __bf16 bf16;
typedef __attribute__((ext_vector_type(16))) __bf16 v16bf;
typedef __attribute__((ext_vector_type(8)))  __bf16 v8bf;
typedef __attribute__((ext_vector_type(8)))  float  v8f;
typedef __attribute__((ext_vector_type(4)))  unsigned int u32x4;
typedef __attribute__((ext_vector_type(4)))  int          i32x4;
typedef __attribute__((ext_vector_type(8)))  int          i32x8;

// ---- fragment helpers (layouts per CDNA5 ISA 7.12.2, wave32) ----

__device__ __forceinline__ v16bf cat16(v8bf lo, v8bf hi) {
    v16bf r;
#pragma unroll
    for (int i = 0; i < 8; ++i) { r[i] = lo[i]; r[i + 8] = hi[i]; }
    return r;
}

// A (16x32 bf16, row-major source, leading dim ld):
// lanes 0-15: row = m0+lane, K 0..7 then 16..23 ; lanes 16-31: K 8..15 then 24..31
__device__ __forceinline__ v16bf load_a16x32(const bf16* __restrict__ base, int ld,
                                             int m0, int kb, int lane) {
    int row = m0 + (lane & 15);
    int kh  = (lane >> 4) << 3;
    const bf16* p = base + (size_t)row * ld + kb + kh;
    return cat16(*(const v8bf*)p, *(const v8bf*)(p + 16));
}

// B (32x16 bf16) from a TRANSPOSED weight matrix WT[N][K]:
// lane = column n0+(lane&15); lanes 0-15: K 0..7 then 16..23 ; lanes 16-31: K 8..15 then 24..31
__device__ __forceinline__ v16bf load_bT32x16(const bf16* __restrict__ baseT, int ld,
                                              int n0, int kb, int lane) {
    int row  = n0 + (lane & 15);
    int koff = kb + ((lane >> 4) << 3);
    const bf16* p = baseT + (size_t)row * ld + koff;
    return cat16(*(const v8bf*)p, *(const v8bf*)(p + 16));
}

__device__ __forceinline__ v8f wmma_bf16(v16bf a, v16bf b, v8f c) {
    return __builtin_amdgcn_wmma_f32_16x16x32_bf16(false, a, false, b, (short)0, c,
                                                   false, false);
}

// ---- Tensor Data Mover: 1-D contiguous copy, global -> LDS ----
// D# per CDNA5 ISA: count=1, type=2 ("image"),
// data_size=8B, tensor_dim0 = tile_dim0 = n8 (8-byte units), tensor/tile_dim1=1.
__device__ __forceinline__ void tdm_load_1d(unsigned lds_addr, const void* gptr,
                                            unsigned n8) {
    unsigned long long ga = (unsigned long long)gptr;
    u32x4 g0;
    g0[0] = 1u;                                              // count=1, user mode
    g0[1] = lds_addr;                                        // lds_addr [63:32]
    g0[2] = (unsigned)ga;                                    // global_addr [95:64]
    g0[3] = (unsigned)((ga >> 32) & 0x01FFFFFFu) | (2u << 30); // addr[56:32] | type=2
    i32x8 g1;
    g1[0] = 0x00030000;                    // wg_mask=0, data_size=3 (8B), no flags
    g1[1] = (int)(n8 << 16);               // tensor_dim0[15:0] in bits [63:48]
    g1[2] = (int)((n8 >> 16) | (1u << 16));// tensor_dim0[31:16] | tensor_dim1=1
    g1[3] = (int)(n8 << 16);               // tensor_dim1 hi=0 | tile_dim0 [127:112]
    g1[4] = 1;                             // tile_dim1=1, tile_dim2=0
    g1[5] = (int)n8;                       // tensor_dim0_stride lo
    g1[6] = 0;
    g1[7] = 0;
    i32x4 z4 = {};
#if defined(__clang_major__) && (__clang_major__ >= 23)
    i32x8 z8 = {};
    __builtin_amdgcn_tensor_load_to_lds(g0, g1, z4, z4, z8, 0);
#else
    __builtin_amdgcn_tensor_load_to_lds(g0, g1, z4, z4, 0);
#endif
}

// ---- prep kernels ----

__global__ void k_init(unsigned* ctr, bf16* h0) {
    size_t i = (size_t)blockIdx.x * blockDim.x + threadIdx.x;
    if (i == 0) *ctr = 0u;
    const size_t n = (size_t)BATCH * UNITS;
    for (size_t j = i; j < n; j += (size_t)gridDim.x * blockDim.x)
        h0[j] = (bf16)0.0f;
}

__global__ void k_cvt(const float* __restrict__ src, bf16* __restrict__ dst, size_t n) {
    for (size_t i = (size_t)blockIdx.x * blockDim.x + threadIdx.x; i < n;
         i += (size_t)gridDim.x * blockDim.x)
        dst[i] = (bf16)src[i];
}

// src is [K][N] fp32 row-major -> dst is [N][K] bf16 row-major
__global__ void k_transpose_cvt(const float* __restrict__ src, bf16* __restrict__ dst,
                                int K, int N) {
    size_t total = (size_t)K * N;
    for (size_t i = (size_t)blockIdx.x * blockDim.x + threadIdx.x; i < total;
         i += (size_t)gridDim.x * blockDim.x) {
        size_t k = i / (size_t)N;
        size_t n = i % (size_t)N;
        dst[n * (size_t)K + k] = (bf16)src[i];
    }
}

// ---- xp = emb[tokens] @ Wx + b   (M=32768 time-major rows, K=256, N=1024) ----

__global__ void __launch_bounds__(256) k_embed_xp(const int* __restrict__ tokens,
                                                  const bf16* __restrict__ embB,
                                                  const bf16* __restrict__ WxT,
                                                  const float* __restrict__ bias,
                                                  bf16* __restrict__ xp) {
    int wave = threadIdx.x >> 5, lane = threadIdx.x & 31;
    int mtile = blockIdx.x >> 3;
    int ntile = ((blockIdx.x & 7) << 3) + wave;
    int m0 = mtile << 4, n0 = ntile << 4;

    int m = m0 + (lane & 15);
    int s = m >> 6, bi = m & 63;
    int tok = tokens[bi * SEQLEN + s];
    const bf16* arow = embB + (size_t)tok * EMB;
    int kh = (lane >> 4) << 3;

    v8f acc = {};
#pragma unroll
    for (int kb = 0; kb < EMB; kb += 32) {
        v16bf a = cat16(*(const v8bf*)(arow + kb + kh),
                        *(const v8bf*)(arow + kb + kh + 16));
        v16bf b = load_bT32x16(WxT, EMB, n0, kb, lane);
        acc = wmma_bf16(a, b, acc);
    }

    int n = n0 + (lane & 15);
    float bn = bias[n];
    int hi8 = (lane >> 4) << 3;
#pragma unroll
    for (int r = 0; r < 8; ++r) {
        int mm = m0 + r + hi8;
        xp[(size_t)mm * UNITS + n] = (bf16)(acc[r] + bn);
    }
}

// ---- persistent recurrent kernel ----
// 16 WGs, each owns 64 output units. LDS: [0,128K) = h_t stage (TDM per step),
// [128K,256K) = this WG's WhT slice (TDM once, resident all 512 steps).

#define LDS_H_BYTES   (BATCH * UNITS * 2)          // 131072
#define LDS_W_OFFSET  LDS_H_BYTES
#define LDS_TOTAL     (2 * LDS_H_BYTES)            // 262144

__global__ void __launch_bounds__(256, 1) k_rnn(const bf16* __restrict__ xp,
                                                const bf16* __restrict__ WhT,
                                                bf16* __restrict__ hbuf,
                                                unsigned* __restrict__ ctr) {
    extern __shared__ char smem_raw[];
    bf16* sh_h = (bf16*)smem_raw;                    // [BATCH][UNITS]
    bf16* sh_w = (bf16*)(smem_raw + LDS_W_OFFSET);   // [64][UNITS] WhT slice
    const int tid = threadIdx.x;
    const int wave = tid >> 5, lane = tid & 31;
    const int wg = blockIdx.x;                       // 0..15
    const unsigned nwg = gridDim.x;

    const int mt  = wave >> 1;                       // 0..3  (M tile)
    const int ntp = (wave & 1) << 1;                 // 0 or 2 (N tile pair)
    const int nl0 = ntp * 16;                        // local cols in sh_w
    const int nl1 = nl0 + 16;
    const int n0 = wg * 64 + nl0;                    // global cols
    const int n1 = n0 + 16;

    // preload this WG's WhT slice (rows n in [wg*64, wg*64+64), contiguous 128KB)
    if (wave == 0)
        tdm_load_1d(LDS_W_OFFSET, WhT + (size_t)wg * 64 * UNITS, LDS_H_BYTES / 8);

    for (int t = 0; t < SEQLEN; ++t) {
        // TDM: stage h_t (128KB contiguous) into LDS; wave0 issues, all sync
        if (wave == 0) {
            tdm_load_1d(0u, hbuf + (size_t)t * BATCH * UNITS, LDS_H_BYTES / 8);
            __builtin_amdgcn_s_wait_tensorcnt(0);    // covers Wh preload too
        }
        __syncthreads();

        if (t + 1 < SEQLEN)   // global_prefetch_b8 for next step's xp tile
            __builtin_prefetch(xp + ((size_t)(t + 1) * BATCH + mt * 16) * UNITS + n0, 0, 1);

        v8f acc0 = {}, acc1 = {};
#pragma unroll 4
        for (int kb = 0; kb < UNITS; kb += 32) {
            v16bf a  = load_a16x32(sh_h, UNITS, mt * 16, kb, lane);
            v16bf b0 = load_bT32x16(sh_w, UNITS, nl0, kb, lane);
            v16bf b1 = load_bT32x16(sh_w, UNITS, nl1, kb, lane);
            acc0 = wmma_bf16(a, b0, acc0);
            acc1 = wmma_bf16(a, b1, acc1);
        }

        // h_{t+1} = tanh(xp_t + h@Wh), written time-major (doubles as hs)
        const int c0 = n0 + (lane & 15), c1 = n1 + (lane & 15);
        const int hi8 = (lane >> 4) << 3;
        const bf16* xr = xp + (size_t)t * BATCH * UNITS;
        bf16* hw = hbuf + (size_t)(t + 1) * BATCH * UNITS;
#pragma unroll
        for (int r = 0; r < 8; ++r) {
            int mm = mt * 16 + r + hi8;
            float v0 = tanhf((float)xr[(size_t)mm * UNITS + c0] + acc0[r]);
            float v1 = tanhf((float)xr[(size_t)mm * UNITS + c1] + acc1[r]);
            hw[(size_t)mm * UNITS + c0] = (bf16)v0;
            hw[(size_t)mm * UNITS + c1] = (bf16)v1;
        }

        // device-scope grid barrier (monotonic counter; reset by k_init each launch)
        __syncthreads();
        if (tid == 0) {
            __hip_atomic_fetch_add(ctr, 1u, __ATOMIC_ACQ_REL, __HIP_MEMORY_SCOPE_AGENT);
            unsigned target = nwg * (unsigned)(t + 1);
            while (__hip_atomic_load(ctr, __ATOMIC_ACQUIRE, __HIP_MEMORY_SCOPE_AGENT) < target)
                __builtin_amdgcn_s_sleep(2);
        }
        __syncthreads();
        __builtin_amdgcn_fence(__ATOMIC_ACQUIRE, "agent");
    }
}

// ---- logits = hs @ Wd + bd   (M=32768, K=1024, N=128) ----

__global__ void __launch_bounds__(256) k_logits(const bf16* __restrict__ hs,
                                                const bf16* __restrict__ WdT,
                                                const float* __restrict__ bd,
                                                float* __restrict__ out) {
    int wave = threadIdx.x >> 5, lane = threadIdx.x & 31;
    int m0 = blockIdx.x << 4;
    int n0 = wave << 4;

    v8f acc = {};
#pragma unroll 4
    for (int kb = 0; kb < UNITS; kb += 32) {
        v16bf a = load_a16x32(hs, UNITS, m0, kb, lane);
        v16bf b = load_bT32x16(WdT, UNITS, n0, kb, lane);
        acc = wmma_bf16(a, b, acc);
    }

    int v = n0 + (lane & 15);
    float bv = bd[v];
    int hi8 = (lane >> 4) << 3;
#pragma unroll
    for (int r = 0; r < 8; ++r) {
        int mm = m0 + r + hi8;                 // time-major row: t = mm>>6, b = mm&63
        int t = mm >> 6, bi = mm & 63;
        out[((size_t)bi * SEQLEN + t) * VOCAB + v] = acc[r] + bv;
    }
}

// ---- host ----

static inline size_t al256(size_t x) { return (x + 255) & ~(size_t)255; }

extern "C" void kernel_launch(void* const* d_in, const int* in_sizes, int n_in,
                              void* d_out, int out_size, void* d_ws, size_t ws_size,
                              hipStream_t stream) {
    (void)in_sizes; (void)n_in; (void)out_size; (void)ws_size;
    const int*   tokens = (const int*)d_in[0];
    const float* emb    = (const float*)d_in[1];
    const float* Wx     = (const float*)d_in[2];
    const float* Wh     = (const float*)d_in[3];
    const float* bias   = (const float*)d_in[4];
    const float* Wd     = (const float*)d_in[5];
    const float* bd     = (const float*)d_in[6];
    float* out = (float*)d_out;

    char* ws = (char*)d_ws;
    size_t off = 0;
    unsigned* ctr = (unsigned*)(ws + off);  off += 256;
    bf16* hbuf = (bf16*)(ws + off);  off += al256((size_t)(SEQLEN + 1) * BATCH * UNITS * 2);
    bf16* xp   = (bf16*)(ws + off);  off += al256((size_t)SEQLEN * BATCH * UNITS * 2);
    bf16* WxT  = (bf16*)(ws + off);  off += al256((size_t)EMB * UNITS * 2);
    bf16* WhT  = (bf16*)(ws + off);  off += al256((size_t)UNITS * UNITS * 2);
    bf16* WdT  = (bf16*)(ws + off);  off += al256((size_t)UNITS * VOCAB * 2);
    bf16* embB = (bf16*)(ws + off);  off += al256((size_t)VOCAB * EMB * 2);

    // prep: zero barrier counter + h0, convert/transpose weights to bf16
    k_init<<<256, 256, 0, stream>>>(ctr, hbuf);
    k_cvt<<<64, 256, 0, stream>>>(emb, embB, (size_t)VOCAB * EMB);
    k_transpose_cvt<<<256, 256, 0, stream>>>(Wx, WxT, EMB, UNITS);
    k_transpose_cvt<<<512, 256, 0, stream>>>(Wh, WhT, UNITS, UNITS);
    k_transpose_cvt<<<128, 256, 0, stream>>>(Wd, WdT, UNITS, VOCAB);

    // xp = emb[tokens] @ Wx + b  (time-major rows)
    k_embed_xp<<<2048 * 8, 256, 0, stream>>>(tokens, embB, WxT, bias, xp);

    // persistent recurrent scan: 16 co-resident WGs, TDM-staged h + resident Wh
    k_rnn<<<16, 256, LDS_TOTAL, stream>>>(xp, WhT, hbuf, ctr);

    // logits = hs @ Wd + bd  (hs = hbuf blocks 1..SEQ)
    k_logits<<<2048, 256, 0, stream>>>(hbuf + (size_t)BATCH * UNITS, WdT, bd, out);
}
